// MultiheadAttentionComp_65996467470544
// MI455X (gfx1250) — matrix operand order repaired and checked by reading, not compiled, not run on hardware
//
#include <hip/hip_runtime.h>

// ---------------- problem constants (reference: S=2048, B=2, E=1024) --------
#define S_LEN  2048
#define BATCH  2
#define NTOK   (S_LEN * BATCH)   // 4096 token rows, r = s*B + b
#define EMB    1024
#define NHEADS 16
#define HDIM   64
#define MLPH   512
#define SIGD   64
#define KEEP   12                // NUM_HEADS - INACTIVE
#define NKB    (S_LEN / 32)      // 32-key blocks per sequence

typedef __bf16 bf16_t;
typedef __bf16 v16bf __attribute__((ext_vector_type(16)));
typedef float  v8f   __attribute__((ext_vector_type(8)));

union AF { v16bf v; bf16_t h[16]; };

static __device__ __forceinline__ v8f wmma_bf16(v16bf a, v16bf b, v8f c) {
  return __builtin_amdgcn_wmma_f32_16x16x32_bf16(false, a, false, b, (short)0, c,
                                                 false, false);
}

static __device__ __forceinline__ void cp8(bf16_t* dst, const bf16_t* src) {
  *reinterpret_cast<uint4*>(dst) = *reinterpret_cast<const uint4*>(src);
}

// A fragment (16 rows x 32 k), row-major bf16 source with row stride K.
// lane -> row = base + (lane&15); element e -> k = (e&7) + (e>=8?16:0) + (lane>=16?8:0)
static __device__ __forceinline__ v16bf loadA_bf(const bf16_t* A, long row, int K,
                                                 int k0, int lane) {
  AF u;
  const bf16_t* p = A + row * (long)K + k0 + ((lane >> 4) << 3);
  cp8(u.h, p); cp8(u.h + 8, p + 16);
  return u.v;
}

// B fragment (32 k x 16 cols) from row-major bf16 W (N x K): B[k][n] = W[n][k].
// lane -> col = base + (lane&15); element e -> k = (e&7) + (e>=8?16:0) + (lane>=16?8:0)
static __device__ __forceinline__ v16bf loadB_bf(const bf16_t* W, long col, int K,
                                                 int k0, int lane) {
  AF u;
  const bf16_t* p = W + col * (long)K + k0 + ((lane >> 4) << 3);
  cp8(u.h, p); cp8(u.h + 8, p + 16);
  return u.v;
}

// ---------------- CDNA5 async global->LDS copy (ASYNCcnt path) --------------
// Generic LDS pointer: low 32 bits are the LDS byte offset (aperture mapping).
static __device__ __forceinline__ void async_cp16(bf16_t* lds, const bf16_t* g) {
  const unsigned lo = (unsigned)(unsigned long long)(uintptr_t)lds;
  const unsigned long long ga = (unsigned long long)(uintptr_t)g;
  asm volatile("global_load_async_to_lds_b128 %0, %1, off"
               :: "v"(lo), "v"(ga) : "memory");
}
static __device__ __forceinline__ void async_wait0() {
  asm volatile("s_wait_asynccnt 0x0" ::: "memory");
}

// ---------------- f32 -> bf16 pre-conversion ---------------------------------
__global__ void cvt_bf16_kernel(const float* __restrict__ src,
                                bf16_t* __restrict__ dst, int n4) {
  const int i = blockIdx.x * blockDim.x + threadIdx.x;
  if (i >= n4) return;
  const float4 f = reinterpret_cast<const float4*>(src)[i];
  union { bf16_t h[4]; uint2 u; } o;
  o.h[0] = (bf16_t)f.x; o.h[1] = (bf16_t)f.y;
  o.h[2] = (bf16_t)f.z; o.h[3] = (bf16_t)f.w;
  reinterpret_cast<uint2*>(dst)[i] = o.u;
}

// ---------------- generic C = A @ W^T + bias (all-bf16 operands) -------------
// 256 threads = 8 waves (4x2), block tile 128x64, wave tile 32x32.
// STORE: 0 = f32, 1 = bf16, 2 = bf16 + relu
template <int STORE>
__global__ __launch_bounds__(256) void gemm_bt(const bf16_t* __restrict__ A,
                                               const bf16_t* __restrict__ W,
                                               const float* __restrict__ bias,
                                               void* __restrict__ Cout,
                                               int M, int N, int K) {
  const int lane = threadIdx.x & 31;
  const int wave = threadIdx.x >> 5;
  const long tM = (long)blockIdx.y * 128 + (wave >> 1) * 32;
  const long tN = (long)blockIdx.x * 64 + (wave & 1) * 32;
  const int rlo = lane & 15;
  v8f acc[2][2] = {};
  for (int k0 = 0; k0 < K; k0 += 32) {
    if (k0 + 32 < K) {
      __builtin_prefetch(A + (tM + rlo) * (long)K + k0 + 32, 0, 1);
      __builtin_prefetch(W + (tN + rlo) * (long)K + k0 + 32, 0, 1);
    }
    v16bf a0 = loadA_bf(A, tM + rlo,      K, k0, lane);
    v16bf a1 = loadA_bf(A, tM + 16 + rlo, K, k0, lane);
    v16bf b0 = loadB_bf(W, tN + rlo,      K, k0, lane);
    v16bf b1 = loadB_bf(W, tN + 16 + rlo, K, k0, lane);
    acc[0][0] = wmma_bf16(a0, b0, acc[0][0]);
    acc[0][1] = wmma_bf16(a0, b1, acc[0][1]);
    acc[1][0] = wmma_bf16(a1, b0, acc[1][0]);
    acc[1][1] = wmma_bf16(a1, b1, acc[1][1]);
  }
  const int hi8 = (lane >> 4) << 3;
#pragma unroll
  for (int si = 0; si < 2; ++si)
#pragma unroll
    for (int sj = 0; sj < 2; ++sj) {
      const long col = tN + sj * 16 + rlo;
      const float bv = bias ? bias[col] : 0.0f;
#pragma unroll
      for (int i = 0; i < 8; ++i) {
        const long row = tM + si * 16 + hi8 + i;
        float v = acc[si][sj][i] + bv;
        if (STORE == 2) v = fmaxf(v, 0.0f);
        if (STORE == 0) ((float*)Cout)[row * N + col] = v;
        else            ((bf16_t*)Cout)[row * N + col] = (bf16_t)v;
      }
    }
}

// ---------------- fused QKV projection ---------------------------------------
// blockIdx.z selects q/k/v weights. q,k -> (B,H,S,D) bf16 (q pre-scaled by
// 1/sqrt(D)=0.125); v -> (B,H,D,S) bf16 (transposed for contiguous PV B-frags).
__global__ __launch_bounds__(256) void qkv_gemm(const bf16_t* __restrict__ X,
    const bf16_t* __restrict__ qw, const float* __restrict__ qb,
    const bf16_t* __restrict__ kw, const float* __restrict__ kb,
    const bf16_t* __restrict__ vw, const float* __restrict__ vb,
    bf16_t* __restrict__ qo, bf16_t* __restrict__ ko, bf16_t* __restrict__ vo) {
  const int lane = threadIdx.x & 31;
  const int wave = threadIdx.x >> 5;
  const int tM = blockIdx.y * 128 + (wave >> 1) * 32;
  const int tN = blockIdx.x * 64 + (wave & 1) * 32;
  const int which = blockIdx.z;
  const bf16_t* W   = (which == 0) ? qw : (which == 1) ? kw : vw;
  const float* bias = (which == 0) ? qb : (which == 1) ? kb : vb;
  const int rlo = lane & 15;
  v8f acc[2][2] = {};
  for (int k0 = 0; k0 < EMB; k0 += 32) {
    v16bf a0 = loadA_bf(X, tM + rlo,      EMB, k0, lane);
    v16bf a1 = loadA_bf(X, tM + 16 + rlo, EMB, k0, lane);
    v16bf b0 = loadB_bf(W, tN + rlo,      EMB, k0, lane);
    v16bf b1 = loadB_bf(W, tN + 16 + rlo, EMB, k0, lane);
    acc[0][0] = wmma_bf16(a0, b0, acc[0][0]);
    acc[0][1] = wmma_bf16(a0, b1, acc[0][1]);
    acc[1][0] = wmma_bf16(a1, b0, acc[1][0]);
    acc[1][1] = wmma_bf16(a1, b1, acc[1][1]);
  }
  const int hi8 = (lane >> 4) << 3;
#pragma unroll
  for (int si = 0; si < 2; ++si)
#pragma unroll
    for (int sj = 0; sj < 2; ++sj) {
      const int col = tN + sj * 16 + rlo;
      const float bv = bias[col];
      const int h = col >> 6, d = col & (HDIM - 1);
#pragma unroll
      for (int i = 0; i < 8; ++i) {
        const int row = tM + si * 16 + hi8 + i;     // token r = s*B + b
        const int s = row >> 1, b = row & 1;        // BATCH == 2
        const float v = acc[si][sj][i] + bv;
        const long bh = (long)(b * NHEADS + h);
        if (which == 0)      qo[(bh * S_LEN + s) * HDIM + d] = (bf16_t)(v * 0.125f);
        else if (which == 1) ko[(bh * S_LEN + s) * HDIM + d] = (bf16_t)v;
        else                 vo[(bh * HDIM + d) * S_LEN + s] = (bf16_t)v;
      }
    }
}

// ---------------- head-signature scores + top-k mask -------------------------
__global__ void head_mask_kernel(const float* __restrict__ t2,   // (NTOK, SIGD)
                                 const float* __restrict__ sig,  // (H, SIGD)
                                 float* __restrict__ mask) {     // (B, H, S)
  const int r = blockIdx.x * blockDim.x + threadIdx.x;
  if (r >= NTOK) return;
  const float* x = t2 + (long)r * SIGD;
  float sc[NHEADS];
#pragma unroll
  for (int h = 0; h < NHEADS; ++h) {
    float a = 0.0f;
    for (int d = 0; d < SIGD; ++d) a += x[d] * sig[h * SIGD + d];
    sc[h] = a;
  }
  float tmp[NHEADS];
#pragma unroll
  for (int h = 0; h < NHEADS; ++h) tmp[h] = sc[h];
  float kth = -__builtin_inff();
  for (int it = 0; it < KEEP; ++it) {       // 12th-largest of 16
    int bi = 0; float bv = -__builtin_inff();
#pragma unroll
    for (int h = 0; h < NHEADS; ++h)
      if (tmp[h] > bv) { bv = tmp[h]; bi = h; }
    kth = bv; tmp[bi] = -__builtin_inff();
  }
  const int s = r >> 1, b = r & 1;          // BATCH == 2
#pragma unroll
  for (int h = 0; h < NHEADS; ++h)
    mask[((long)(b * NHEADS + h)) * S_LEN + s] = (sc[h] >= kth) ? 1.0f : 0.0f;
}

// ---------------- flash attention + head gating ------------------------------
// grid (S/16/8, B*H); 8 waves/block, each wave owns 16 query rows.
// K/V tiles are double-buffered in LDS via global_load_async_to_lds_b128.
__global__ __launch_bounds__(256) void attn_fwd(const bf16_t* __restrict__ q,
                                                const bf16_t* __restrict__ k,
                                                const bf16_t* __restrict__ vt,
                                                const float* __restrict__ mask,
                                                bf16_t* __restrict__ og) {
  __shared__ __align__(16) bf16_t ktile[2][32 * HDIM];  // [key][d]   4KB each
  __shared__ __align__(16) bf16_t vtile[2][HDIM * 32];  // [d][key]   4KB each
  __shared__ __align__(16) bf16_t pt[8][16 * 32];       // per-wave P bounce
  const int tid  = threadIdx.x;
  const int lane = tid & 31;
  const int wave = tid >> 5;
  const int bh = blockIdx.y;                // b*H + h
  const int b = bh >> 4, h = bh & 15;
  const int qbase = (blockIdx.x * 8 + wave) * 16;
  const bf16_t* qh = q  + (long)bh * S_LEN * HDIM;
  const bf16_t* kh = k  + (long)bh * S_LEN * HDIM;
  const bf16_t* vh = vt + (long)bh * HDIM * S_LEN;
  const int rlo = lane & 15;
  const int hi8 = (lane >> 4) << 3;
  const int hi16 = (lane >> 4) << 4;

  // cooperative staging coords: 256 threads x 16B cover each 4KB tile
  const int kr = tid >> 3, kc = (tid & 7) * 8;   // ktile: key row, d chunk
  const int vr = tid >> 2, vc = (tid & 3) * 8;   // vtile: d row, key chunk
  auto stage = [&](int bi, int kbase) {
    async_cp16(&ktile[bi][kr * HDIM + kc], kh + (long)(kbase + kr) * HDIM + kc);
    async_cp16(&vtile[bi][vr * 32 + vc],   vh + (long)vr * S_LEN + kbase + vc);
  };

  // Q fragments (held in registers for the whole pass)
  v16bf aq[2];
#pragma unroll
  for (int t = 0; t < 2; ++t) {
    AF u;
    const bf16_t* p = qh + (long)(qbase + rlo) * HDIM + t * 32 + hi8;
    cp8(u.h, p); cp8(u.h + 8, p + 16);
    aq[t] = u.v;
  }

  float mrow[8], lrow[8];
#pragma unroll
  for (int i = 0; i < 8; ++i) { mrow[i] = -__builtin_inff(); lrow[i] = 0.0f; }
  v8f accO[4] = {};

  stage(0, 0);
  async_wait0();
  __syncthreads();

  for (int kb = 0; kb < NKB; ++kb) {
    const int cur = kb & 1;
    if (kb + 1 < NKB) stage(cur ^ 1, (kb + 1) * 32);   // overlap copy/compute
    const bf16_t* kbuf = ktile[cur];
    const bf16_t* vbuf = vtile[cur];
    const int kbase = kb * 32;
    (void)kbase;

    // scores: 16 queries x 32 keys, K-dim = d (64) -> 4 wmma (B-frags from LDS)
    v8f sc[2] = {};
#pragma unroll
    for (int u2 = 0; u2 < 2; ++u2)
#pragma unroll
      for (int t = 0; t < 2; ++t) {
        AF ub;
        const bf16_t* p = kbuf + (u2 * 16 + rlo) * HDIM + t * 32 + hi8;
        cp8(ub.h, p); cp8(ub.h + 8, p + 16);
        sc[u2] = wmma_bf16(aq[t], ub.v, sc[u2]);
      }

    // online softmax (row i lives in element i across the 16-lane half)
    float bm[8];
#pragma unroll
    for (int i = 0; i < 8; ++i) bm[i] = fmaxf(sc[0][i], sc[1][i]);
#pragma unroll
    for (int d = 1; d < 16; d <<= 1)
#pragma unroll
      for (int i = 0; i < 8; ++i) bm[i] = fmaxf(bm[i], __shfl_xor(bm[i], d, 32));
    float f[8], ps[8];
#pragma unroll
    for (int i = 0; i < 8; ++i) {
      const float mn = fmaxf(mrow[i], bm[i]);
      f[i] = __expf(mrow[i] - mn);
      mrow[i] = mn;
      ps[i] = 0.0f;
    }
#pragma unroll
    for (int u2 = 0; u2 < 2; ++u2)
#pragma unroll
      for (int i = 0; i < 8; ++i) {
        const float p = __expf(sc[u2][i] - mrow[i]);
        sc[u2][i] = p;
        ps[i] += p;
      }
#pragma unroll
    for (int d = 1; d < 16; d <<= 1)
#pragma unroll
      for (int i = 0; i < 8; ++i) ps[i] += __shfl_xor(ps[i], d, 32);
#pragma unroll
    for (int i = 0; i < 8; ++i) lrow[i] = lrow[i] * f[i] + ps[i];
#pragma unroll
    for (int dt = 0; dt < 4; ++dt)
#pragma unroll
      for (int i = 0; i < 8; ++i) accO[dt][i] *= f[i];

    // re-layout P (C-frag) -> A-frag through per-wave LDS tile
#pragma unroll
    for (int u2 = 0; u2 < 2; ++u2)
#pragma unroll
      for (int i = 0; i < 8; ++i)
        pt[wave][(hi8 + i) * 32 + u2 * 16 + rlo] = (bf16_t)sc[u2][i];
    __syncthreads();
    AF up;
    const bf16_t* pr = &pt[wave][rlo * 32 + hi8];
    cp8(up.h, pr); cp8(up.h + 8, pr + 16);

    // P @ V : N-dim = d (64 -> 4 tiles), K-dim = 32 keys (B-frags from LDS)
#pragma unroll
    for (int dt = 0; dt < 4; ++dt) {
      AF uv;
      const bf16_t* p = vbuf + (dt * 16 + rlo) * 32 + hi8;
      cp8(uv.h, p); cp8(uv.h + 8, p + 16);
      accO[dt] = wmma_bf16(up.v, uv.v, accO[dt]);
    }

    async_wait0();          // next tiles landed (and this wave's issue drained)
    __syncthreads();        // everyone done reading current buffers
  }

  // normalize, gate by head mask, store as rows of (S*B, E) bf16
  float inv[8], mv[8];
#pragma unroll
  for (int i = 0; i < 8; ++i) {
    inv[i] = 1.0f / lrow[i];
    mv[i] = mask[(long)bh * S_LEN + qbase + hi8 + i];
  }
#pragma unroll
  for (int dt = 0; dt < 4; ++dt)
#pragma unroll
    for (int i = 0; i < 8; ++i) {
      const long s = qbase + hi8 + i;
      const long r = s * BATCH + b;
      const long col = (long)h * HDIM + dt * 16 + rlo;
      og[r * EMB + col] = (bf16_t)(accO[dt][i] * inv[i] * mv[i]);
    }
}

// ---------------- launcher ---------------------------------------------------
extern "C" void kernel_launch(void* const* d_in, const int* in_sizes, int n_in,
                              void* d_out, int out_size, void* d_ws, size_t ws_size,
                              hipStream_t stream) {
  (void)in_sizes; (void)n_in; (void)out_size; (void)ws_size;
  const float* query  = (const float*)d_in[0];
  const float* q_w    = (const float*)d_in[1];
  const float* q_b    = (const float*)d_in[2];
  const float* k_w    = (const float*)d_in[3];
  const float* k_b    = (const float*)d_in[4];
  const float* v_w    = (const float*)d_in[5];
  const float* v_b    = (const float*)d_in[6];
  const float* out_w  = (const float*)d_in[7];
  const float* out_b  = (const float*)d_in[8];
  const float* inf1_w = (const float*)d_in[9];
  const float* inf1_b = (const float*)d_in[10];
  const float* inf2_w = (const float*)d_in[11];
  const float* inf2_b = (const float*)d_in[12];
  const float* hsig   = (const float*)d_in[13];

  char* ws = (char*)d_ws;                                  // ~55.1 MiB used
  bf16_t* q_ws = (bf16_t*)(ws);                            // 8 MiB (B,H,S,D)
  bf16_t* k_ws = (bf16_t*)(ws + (8ull  << 20));            // 8 MiB (B,H,S,D)
  bf16_t* v_ws = (bf16_t*)(ws + (16ull << 20));            // 8 MiB (B,H,D,S)
  bf16_t* og   = (bf16_t*)(ws + (24ull << 20));            // 8 MiB (S*B, E)
  bf16_t* t1   = (bf16_t*)(ws + (32ull << 20));            // 4 MiB (NTOK, 512)
  float*  t2   = (float*) (ws + (36ull << 20));            // 1 MiB (NTOK, 64)
  float*  hmsk = (float*) (ws + (37ull << 20));            // 256 KiB (B,H,S)
  bf16_t* xbf  = (bf16_t*)(ws + (38ull << 20));            // 8 MiB query bf16
  bf16_t* wqb  = (bf16_t*)(ws + (46ull << 20));            // 2 MiB
  bf16_t* wkb  = (bf16_t*)(ws + (48ull << 20));            // 2 MiB
  bf16_t* wvb  = (bf16_t*)(ws + (50ull << 20));            // 2 MiB
  bf16_t* wob  = (bf16_t*)(ws + (52ull << 20));            // 2 MiB
  bf16_t* w1b  = (bf16_t*)(ws + (54ull << 20));            // 1 MiB
  bf16_t* w2b  = (bf16_t*)(ws + (55ull << 20));            // 64 KiB

  const dim3 blk(256);
  auto cvt = [&](const float* s, bf16_t* d, int n) {
    const int n4 = n >> 2;
    cvt_bf16_kernel<<<dim3((n4 + 255) / 256), blk, 0, stream>>>(s, d, n4);
  };
  // one-time f32 -> bf16 operand conversion (keeps GEMM loops cvt-free)
  cvt(query,  xbf, NTOK * EMB);
  cvt(q_w,    wqb, EMB * EMB);
  cvt(k_w,    wkb, EMB * EMB);
  cvt(v_w,    wvb, EMB * EMB);
  cvt(out_w,  wob, EMB * EMB);
  cvt(inf1_w, w1b, MLPH * EMB);
  cvt(inf2_w, w2b, SIGD * MLPH);

  // head-mask MLP: relu(x @ W1^T + b1) @ W2^T + b2
  gemm_bt<2><<<dim3(MLPH / 64, NTOK / 128), blk, 0, stream>>>(
      xbf, w1b, inf1_b, t1, NTOK, MLPH, EMB);
  gemm_bt<0><<<dim3(SIGD / 64, NTOK / 128), blk, 0, stream>>>(
      t1, w2b, inf2_b, t2, NTOK, SIGD, MLPH);
  head_mask_kernel<<<dim3(NTOK / 256), blk, 0, stream>>>(t2, hsig, hmsk);
  // fused QKV projections
  qkv_gemm<<<dim3(EMB / 64, NTOK / 128, 3), blk, 0, stream>>>(
      xbf, wqb, q_b, wkb, k_b, wvb, v_b, q_ws, k_ws, v_ws);
  // attention + gating
  attn_fwd<<<dim3(S_LEN / 16 / 8, BATCH * NHEADS), blk, 0, stream>>>(
      q_ws, k_ws, v_ws, hmsk, og);
  // output projection -> d_out (S, B, E) f32
  gemm_bt<0><<<dim3(EMB / 64, NTOK / 128), blk, 0, stream>>>(
      og, wob, out_b, (float*)d_out, NTOK, EMB, EMB);
}
